// MultiHeadCausalSelfAttention_17085379903948
// MI455X (gfx1250) — matrix operand order repaired and checked by reading, not compiled, not run on hardware
//
#include <hip/hip_runtime.h>

// ---------------------------------------------------------------------------
// Types for CDNA5 WMMA (wave32): v_wmma_f32_16x16x32_bf16
// ---------------------------------------------------------------------------
typedef __attribute__((ext_vector_type(16))) __bf16 v16bf;
typedef __attribute__((ext_vector_type(8)))  __bf16 v8bf;
typedef __attribute__((ext_vector_type(8)))  float  v8f;

constexpr int D_MODEL = 768;
constexpr int N_HEADS = 12;
constexpr int D_K     = 64;
constexpr int BATCH   = 2;
constexpr int SEQ     = 4096;
constexpr int BT      = BATCH * SEQ;   // 8192 rows total
constexpr int E3      = 3 * D_MODEL;   // 2304 qkv outputs
constexpr int KSTEPS  = D_MODEL / 32;  // 24 k-iterations

__device__ __forceinline__ v8bf ld8(const __bf16* p) { return *(const v8bf*)p; }

__device__ __forceinline__ v16bf cat8(v8bf a, v8bf b) {
    v16bf r;
#pragma unroll
    for (int i = 0; i < 8; ++i) { r[i] = a[i]; r[i + 8] = b[i]; }
    return r;
}

__device__ __forceinline__ v8bf scale8(v8bf a, float s) {
#pragma unroll
    for (int i = 0; i < 8; ++i) a[i] = (__bf16)((float)a[i] * s);
    return a;
}

__device__ __forceinline__ unsigned short bfbits(float f) {
    __bf16 b = (__bf16)f;
    unsigned short u;
    __builtin_memcpy(&u, &b, 2);
    return u;
}

__device__ __forceinline__ v8f wmma_bf16(v16bf a, v16bf b, v8f c) {
    return __builtin_amdgcn_wmma_f32_16x16x32_bf16(
        /*neg_a=*/false, a, /*neg_b=*/false, b,
        /*c_mod=*/(short)0, c, /*reuse_a=*/false, /*reuse_b=*/false);
}

// CDNA5 async copy: per-lane 16B global -> LDS, tracked on ASYNCcnt.
// GVS addressing: mem = SGPR_base + VGPR_offset(i32); LDS dest = VDST vgpr.
__device__ __forceinline__ void async_b128_to_lds(unsigned lds_addr,
                                                  unsigned gbyte_off,
                                                  const void* sbase) {
    asm volatile("global_load_async_to_lds_b128 %0, %1, %2"
                 :: "v"(lds_addr), "v"(gbyte_off), "s"(sbase)
                 : "memory");
}

// ---------------------------------------------------------------------------
// K0: fp32 -> bf16 conversion (grid-stride)
// ---------------------------------------------------------------------------
__global__ void cvt_f32_bf16(const float* __restrict__ in,
                             __bf16* __restrict__ out, int n) {
    for (int i = blockIdx.x * blockDim.x + threadIdx.x; i < n;
         i += gridDim.x * blockDim.x)
        out[i] = (__bf16)in[i];
}

// ---------------------------------------------------------------------------
// K1: QKV projection  qkv[t,e] = sum_d x[t,d] * w_qkv[e,d]
// Block = 256 threads = 8 waves. Block tile: 256 M-rows x 64 N-cols.
// Wave tile: 32x64 (8 wmma / k-step). The shared 64x32 weight tile is
// staged in LDS via double-buffered GLOBAL_LOAD_ASYNC_TO_LDS_B128.
// Results scattered to Q [bh,t,64], K [bh,t,64], Vt [bh,64,t] (bf16).
// ---------------------------------------------------------------------------
__global__ __launch_bounds__(256) void qkv_gemm(
    const __bf16* __restrict__ xb, const __bf16* __restrict__ wq,
    __bf16* __restrict__ Qo, __bf16* __restrict__ Ko, __bf16* __restrict__ Vt) {
    __shared__ __align__(16) unsigned short bsh[2][64 * 32];  // 2 x 4KB

    const int tid  = threadIdx.x;
    const int lane = tid & 31;
    const int wid  = tid >> 5;
    const int nstrips = E3 / 64;                  // 36
    const int e0 = (blockIdx.x % nstrips) * 64;
    const int m0 = (blockIdx.x / nstrips) * 256 + wid * 32;
    const int lr = lane & 15, lh = lane >> 4;

    // async staging map: thread -> (row 0..63, 16B chunk 0..3) of 64x32 tile
    const int crow = tid >> 2, cchk = tid & 3;
    const unsigned gbase =
        (unsigned)(((e0 + crow) * D_MODEL + cchk * 8) * 2);   // bytes
    const unsigned lelem = (unsigned)(crow * 32 + cchk * 8);  // elements
    const unsigned lds0 = (unsigned)(uintptr_t)(&bsh[0][0]) + lelem * 2;
    const unsigned lds1 = (unsigned)(uintptr_t)(&bsh[1][0]) + lelem * 2;

    async_b128_to_lds(lds0, gbase, wq);  // prefetch k-step 0

    v8f acc[8] = {v8f{}, v8f{}, v8f{}, v8f{}, v8f{}, v8f{}, v8f{}, v8f{}};
    const __bf16* a0 = xb + (size_t)(m0 + lr) * D_MODEL + lh * 8;
    const __bf16* a1 = xb + (size_t)(m0 + 16 + lr) * D_MODEL + lh * 8;

    for (int ks = 0; ks < KSTEPS; ++ks) {
        const int k0 = ks * 32;
        if (ks + 1 < KSTEPS) {
            async_b128_to_lds(((ks + 1) & 1) ? lds1 : lds0,
                              gbase + (unsigned)((k0 + 32) * 2), wq);
            asm volatile("s_wait_asynccnt 1" ::: "memory");  // buf ks ready
        } else {
            asm volatile("s_wait_asynccnt 0" ::: "memory");
        }
        __syncthreads();

        const unsigned short* bt = &bsh[ks & 1][0];
        v16bf A0 = cat8(ld8(a0 + k0), ld8(a0 + k0 + 16));
        v16bf A1 = cat8(ld8(a1 + k0), ld8(a1 + k0 + 16));
#pragma unroll
        for (int s = 0; s < 4; ++s) {
            const unsigned short* bp = bt + (s * 16 + lr) * 32 + lh * 8;
            v16bf B = cat8(*(const v8bf*)bp, *(const v8bf*)(bp + 16));
            acc[s]     = wmma_bf16(A0, B, acc[s]);
            acc[4 + s] = wmma_bf16(A1, B, acc[4 + s]);
        }
        __syncthreads();
    }

#pragma unroll
    for (int s = 0; s < 4; ++s) {
        const int e    = e0 + s * 16 + lr;
        const int part = e / D_MODEL;  // 0=Q 1=K 2=V
        const int eh   = e % D_MODEL;
        const int h    = eh / D_K;
        const int dk   = eh % D_K;
#pragma unroll
        for (int mm = 0; mm < 2; ++mm) {
#pragma unroll
            for (int r = 0; r < 8; ++r) {
                const int m = m0 + mm * 16 + lh * 8 + r;
                const int b = m / SEQ, t = m % SEQ;
                const int bh = b * N_HEADS + h;
                __bf16 v = (__bf16)acc[mm * 4 + s][r];
                if (part == 0)
                    Qo[((size_t)bh * SEQ + t) * D_K + dk] = v;
                else if (part == 1)
                    Ko[((size_t)bh * SEQ + t) * D_K + dk] = v;
                else
                    Vt[((size_t)bh * D_K + dk) * SEQ + t] = v;  // transposed
            }
        }
    }
}

// ---------------------------------------------------------------------------
// K2: Flash attention, causal. One wave = 16 queries of one (b,h).
//     Key loop step 32: S = Q*K^T (4 wmma), online softmax, P via LDS
//     (C-layout -> A-layout), O += P*V (4 wmma, V pre-transposed).
// ---------------------------------------------------------------------------
__global__ __launch_bounds__(128) void flash_attn(
    const __bf16* __restrict__ Qo, const __bf16* __restrict__ Ko,
    const __bf16* __restrict__ Vt, __bf16* __restrict__ Ob) {
    __shared__ unsigned short psh[4][16 * 32];  // per-wave 1KB P staging

    const int lane = threadIdx.x & 31;
    const int wid  = threadIdx.x >> 5;
    const int gw   = blockIdx.x * 4 + wid;
    const int qt   = gw & 255;           // T/16 = 256 q-tiles
    const int bh   = gw >> 8;            // b*H + h
    const int q0   = qt * 16;
    const int lr   = lane & 15, lh = lane >> 4;

    // Q tile, A-layout, pre-scaled by 1/sqrt(d_k)
    const float qscale = 0.125f;
    const __bf16* qrow = Qo + ((size_t)bh * SEQ + q0 + lr) * D_K + lh * 8;
    v16bf Aq[2];
#pragma unroll
    for (int c = 0; c < 2; ++c)
        Aq[c] = cat8(scale8(ld8(qrow + c * 32), qscale),
                     scale8(ld8(qrow + c * 32 + 16), qscale));

    float mrow[8], lrow[8];
    v8f oacc[4] = {v8f{}, v8f{}, v8f{}, v8f{}};
#pragma unroll
    for (int r = 0; r < 8; ++r) { mrow[r] = -3.0e38f; lrow[r] = 0.0f; }

    for (int kt = 0; kt < q0 + 16; kt += 32) {
        // ---- S = Q * K^T  (16 x 32 scores) ----
        v8f s[2] = {v8f{}, v8f{}};
#pragma unroll
        for (int sub = 0; sub < 2; ++sub) {
            const __bf16* krow =
                Ko + ((size_t)bh * SEQ + kt + sub * 16 + lr) * D_K + lh * 8;
#pragma unroll
            for (int c = 0; c < 2; ++c) {
                v16bf B = cat8(ld8(krow + c * 32), ld8(krow + c * 32 + 16));
                s[sub] = wmma_bf16(Aq[c], B, s[sub]);
            }
        }

        // ---- causal mask + online softmax ----
        float alpha[8];
#pragma unroll
        for (int r = 0; r < 8; ++r) {
            const int qg = q0 + lh * 8 + r;
            float s0 = s[0][r], s1 = s[1][r];
            if (kt + lr > qg)      s0 = -3.0e38f;
            if (kt + 16 + lr > qg) s1 = -3.0e38f;
            float rm = fmaxf(s0, s1);
            rm = fmaxf(rm, __shfl_xor(rm, 1));
            rm = fmaxf(rm, __shfl_xor(rm, 2));
            rm = fmaxf(rm, __shfl_xor(rm, 4));
            rm = fmaxf(rm, __shfl_xor(rm, 8));
            const float mn = fmaxf(mrow[r], rm);
            const float al = __expf(mrow[r] - mn);
            mrow[r] = mn;
            const float p0 = __expf(s0 - mn);
            const float p1 = __expf(s1 - mn);
            float rs = p0 + p1;
            rs += __shfl_xor(rs, 1);
            rs += __shfl_xor(rs, 2);
            rs += __shfl_xor(rs, 4);
            rs += __shfl_xor(rs, 8);
            lrow[r] = lrow[r] * al + rs;
            alpha[r] = al;
            const int row = lh * 8 + r;          // C-layout -> row-major LDS
            psh[wid][row * 32 + lr]      = bfbits(p0);
            psh[wid][row * 32 + 16 + lr] = bfbits(p1);
        }
#pragma unroll
        for (int nt = 0; nt < 4; ++nt)
#pragma unroll
            for (int r = 0; r < 8; ++r) oacc[nt][r] *= alpha[r];

        // wave-internal LDS visibility (same wave wrote, same wave reads)
        __builtin_amdgcn_wave_barrier();
        asm volatile("s_wait_dscnt 0" ::: "memory");

        // ---- reload P in A-layout (16 x 32) ----
        const unsigned short* pb = &psh[wid][lr * 32 + lh * 8];
        v16bf Ap = cat8(*(const v8bf*)pb, *(const v8bf*)(pb + 16));
        __builtin_amdgcn_wave_barrier();
        asm volatile("" ::: "memory");

        // ---- O += P * V   (V transposed: contiguous along keys) ----
#pragma unroll
        for (int nt = 0; nt < 4; ++nt) {
            const __bf16* vrow =
                Vt + ((size_t)bh * D_K + nt * 16 + lr) * SEQ + kt + lh * 8;
            v16bf B = cat8(ld8(vrow), ld8(vrow + 16));
            oacc[nt] = wmma_bf16(Ap, B, oacc[nt]);
        }
    }

    // ---- finalize: divide by l, store bf16 to O [b,t, h*64+dk] ----
    const int b = bh / N_HEADS, h = bh % N_HEADS;
#pragma unroll
    for (int r = 0; r < 8; ++r) {
        const float inv = 1.0f / lrow[r];
        const int t = q0 + lh * 8 + r;
#pragma unroll
        for (int nt = 0; nt < 4; ++nt)
            Ob[((size_t)(b * SEQ + t)) * D_MODEL + h * D_K + nt * 16 + lr] =
                (__bf16)(oacc[nt][r] * inv);
    }
}

// ---------------------------------------------------------------------------
// K3: output projection  y[t,e] = sum_d o[t,d] * w_out[e,d]  (f32 out)
// Same block structure as K1: 256x64 block tile, async LDS weight staging.
// ---------------------------------------------------------------------------
__global__ __launch_bounds__(256) void out_gemm(
    const __bf16* __restrict__ Ob, const __bf16* __restrict__ wo,
    float* __restrict__ y) {
    __shared__ __align__(16) unsigned short bsh[2][64 * 32];

    const int tid  = threadIdx.x;
    const int lane = tid & 31;
    const int wid  = tid >> 5;
    const int nstrips = D_MODEL / 64;             // 12
    const int e0 = (blockIdx.x % nstrips) * 64;
    const int m0 = (blockIdx.x / nstrips) * 256 + wid * 32;
    const int lr = lane & 15, lh = lane >> 4;

    const int crow = tid >> 2, cchk = tid & 3;
    const unsigned gbase =
        (unsigned)(((e0 + crow) * D_MODEL + cchk * 8) * 2);
    const unsigned lelem = (unsigned)(crow * 32 + cchk * 8);
    const unsigned lds0 = (unsigned)(uintptr_t)(&bsh[0][0]) + lelem * 2;
    const unsigned lds1 = (unsigned)(uintptr_t)(&bsh[1][0]) + lelem * 2;

    async_b128_to_lds(lds0, gbase, wo);

    v8f acc[8] = {v8f{}, v8f{}, v8f{}, v8f{}, v8f{}, v8f{}, v8f{}, v8f{}};
    const __bf16* a0 = Ob + (size_t)(m0 + lr) * D_MODEL + lh * 8;
    const __bf16* a1 = Ob + (size_t)(m0 + 16 + lr) * D_MODEL + lh * 8;

    for (int ks = 0; ks < KSTEPS; ++ks) {
        const int k0 = ks * 32;
        if (ks + 1 < KSTEPS) {
            async_b128_to_lds(((ks + 1) & 1) ? lds1 : lds0,
                              gbase + (unsigned)((k0 + 32) * 2), wo);
            asm volatile("s_wait_asynccnt 1" ::: "memory");
        } else {
            asm volatile("s_wait_asynccnt 0" ::: "memory");
        }
        __syncthreads();

        const unsigned short* bt = &bsh[ks & 1][0];
        v16bf A0 = cat8(ld8(a0 + k0), ld8(a0 + k0 + 16));
        v16bf A1 = cat8(ld8(a1 + k0), ld8(a1 + k0 + 16));
#pragma unroll
        for (int s = 0; s < 4; ++s) {
            const unsigned short* bp = bt + (s * 16 + lr) * 32 + lh * 8;
            v16bf B = cat8(*(const v8bf*)bp, *(const v8bf*)(bp + 16));
            acc[s]     = wmma_bf16(A0, B, acc[s]);
            acc[4 + s] = wmma_bf16(A1, B, acc[4 + s]);
        }
        __syncthreads();
    }

#pragma unroll
    for (int s = 0; s < 4; ++s)
#pragma unroll
        for (int mm = 0; mm < 2; ++mm)
#pragma unroll
            for (int r = 0; r < 8; ++r)
                y[(size_t)(m0 + mm * 16 + lh * 8 + r) * D_MODEL + e0 +
                  s * 16 + lr] = acc[mm * 4 + s][r];
}

// ---------------------------------------------------------------------------
extern "C" void kernel_launch(void* const* d_in, const int* in_sizes, int n_in,
                              void* d_out, int out_size, void* d_ws,
                              size_t ws_size, hipStream_t stream) {
    const float* x     = (const float*)d_in[0];  // [2,4096,768]
    const float* w_qkv = (const float*)d_in[1];  // [2304,768]
    const float* w_out = (const float*)d_in[2];  // [768,768]
    float* y = (float*)d_out;

    char* w = (char*)d_ws;
    __bf16* xb  = (__bf16*)w; w += (size_t)BT * D_MODEL * 2;
    __bf16* wqb = (__bf16*)w; w += (size_t)E3 * D_MODEL * 2;
    __bf16* wob = (__bf16*)w; w += (size_t)D_MODEL * D_MODEL * 2;
    __bf16* Qb  = (__bf16*)w; w += (size_t)BT * D_MODEL * 2;
    __bf16* Kb  = (__bf16*)w; w += (size_t)BT * D_MODEL * 2;
    __bf16* Vtb = (__bf16*)w; w += (size_t)BT * D_MODEL * 2;
    __bf16* Ob  = (__bf16*)w; w += (size_t)BT * D_MODEL * 2;

    cvt_f32_bf16<<<2048, 256, 0, stream>>>(x, xb, BT * D_MODEL);
    cvt_f32_bf16<<<2048, 256, 0, stream>>>(w_qkv, wqb, E3 * D_MODEL);
    cvt_f32_bf16<<<1024, 256, 0, stream>>>(w_out, wob, D_MODEL * D_MODEL);

    // (BT/256) * (2304/64) = 32*36 = 1152 blocks of 8 waves
    qkv_gemm<<<1152, 256, 0, stream>>>(xb, wqb, Qb, Kb, Vtb);

    // B*H*(T/16) = 6144 waves, 4 waves/block
    flash_attn<<<1536, 128, 0, stream>>>(Qb, Kb, Vtb, Ob);

    // (BT/256) * (768/64) = 32*12 = 384 blocks
    out_gemm<<<384, 256, 0, stream>>>(Ob, wob, y);
}
